// GINEConv_74500502716622
// MI455X (gfx1250) — matrix operand order repaired and checked by reading, not compiled, not run on hardware
//
#include <hip/hip_runtime.h>

#define DFEAT 128
#define DPAD  136   // 136 halves = 272B rows -> every 16-half-aligned fragment is 16B aligned (ds_load_b128)

typedef __attribute__((ext_vector_type(16))) _Float16 v16h;
typedef __attribute__((ext_vector_type(8)))  _Float16 v8h;
typedef __attribute__((ext_vector_type(8)))  float    v8f;
typedef __attribute__((ext_vector_type(4)))  float    vf4;

// ---------------------------------------------------------------------------
// Kernel 1: agg = x   (folds the (1+eps)*x term, eps = 0)
// ---------------------------------------------------------------------------
__global__ __launch_bounds__(256) void gine_init_kernel(
    const float4* __restrict__ x, float4* __restrict__ agg, int n4)
{
    int i = blockIdx.x * 256 + threadIdx.x;
    if (i < n4) agg[i] = x[i];
}

// ---------------------------------------------------------------------------
// Kernel 2: per-edge  agg[dst] += relu(x[src] + edge_attr[e])
// One wave32 per edge; each lane owns one float4 (16B) of the 512B row.
// edge_attr is streamed once (819MB) -> non-temporal, keep x/agg L2-resident.
// ---------------------------------------------------------------------------
__global__ __launch_bounds__(256) void gine_edge_kernel(
    const float* __restrict__ x,
    const int*   __restrict__ ei,   // [2, E]: src row then dst row
    const float* __restrict__ ea,   // [E, 128]
    float*       __restrict__ agg,  // [N, 128]
    int nedges)
{
    const int lane = threadIdx.x & 31;
    const int wave = threadIdx.x >> 5;
    const int e = blockIdx.x * 8 + wave;
    if (e >= nedges) return;

    const int src = ei[e];
    const int dst = ei[nedges + e];

    const vf4 xv = *(const vf4*)(x + (size_t)src * DFEAT + lane * 4);
    const vf4 av = __builtin_nontemporal_load(
        (const vf4*)(ea + (size_t)e * DFEAT + lane * 4));

    vf4 m;
    m.x = fmaxf(xv.x + av.x, 0.0f);
    m.y = fmaxf(xv.y + av.y, 0.0f);
    m.z = fmaxf(xv.z + av.z, 0.0f);
    m.w = fmaxf(xv.w + av.w, 0.0f);

    float* dp = agg + (size_t)dst * DFEAT + lane * 4;
    unsafeAtomicAdd(dp + 0, m.x);   // native global_atomic_add_f32 (L2-resident)
    unsafeAtomicAdd(dp + 1, m.y);
    unsafeAtomicAdd(dp + 2, m.z);
    unsafeAtomicAdd(dp + 3, m.w);
}

// ---------------------------------------------------------------------------
// Kernel 3: out = relu(agg @ W1 + b1) @ W2 + b2   (fused, WMMA f16 -> f32 acc)
// Block = 256 threads = 8 waves; 16 rows per block; wave w owns cols [16w,16w+16).
// ---------------------------------------------------------------------------
__global__ __launch_bounds__(256) void gine_mlp_kernel(
    const float* __restrict__ agg,
    const float* __restrict__ w1, const float* __restrict__ b1,
    const float* __restrict__ w2, const float* __restrict__ b2,
    float* __restrict__ out, int nodes)
{
    __shared__ __align__(16) _Float16 sW[DFEAT][DPAD];  // transposed: sW[n][k] = W[k][n]
    __shared__ __align__(16) _Float16 sH[16][DPAD];     // input tile (f16)
    __shared__ __align__(16) _Float16 sT[16][DPAD];     // hidden tile (f16)

    const int tid  = threadIdx.x;
    const int lane = tid & 31;
    const int wave = tid >> 5;          // 0..7
    const int m0   = blockIdx.x * 16;

    // ---- stage H tile + W1 (transposed) into LDS -------------------------
    for (int i = tid; i < 16 * DFEAT; i += 256) {
        int r = i >> 7, c = i & 127;
        float v = (m0 + r < nodes) ? agg[(size_t)(m0 + r) * DFEAT + c] : 0.0f;
        sH[r][c] = (_Float16)v;
    }
    for (int i = tid; i < DFEAT * DFEAT; i += 256) {
        int k = i >> 7, n = i & 127;
        sW[n][k] = (_Float16)w1[i];
    }
    __syncthreads();

    const int mrow = lane & 15;         // A-row / D-column index within tile
    const int hi   = lane >> 4;         // lane-half select
    const int n0   = wave * 16;
    const int ncol = n0 + mrow;         // output column owned by this lane

    union Frag { v16h v; v8h h[2]; };

    // ---- stage 1: T = relu(H @ W1 + b1) ----------------------------------
    v8f acc = {};
#pragma unroll
    for (int kb = 0; kb < 4; ++kb) {
        Frag a, b;
        const int abase = kb * 32 + hi * 8;     // A: K = 16*(i/8) + 8*hi + i%8
        a.h[0] = *(const v8h*)&sH[mrow][abase];        // 16B aligned -> ds_load_b128
        a.h[1] = *(const v8h*)&sH[mrow][abase + 16];
        const int bbase = kb * 32 + hi * 8;     // B: K = 16*(i/8) + 8*hi + i%8, col = ncol
        b.h[0] = *(const v8h*)&sW[ncol][bbase];
        b.h[1] = *(const v8h*)&sW[ncol][bbase + 16];
        acc = __builtin_amdgcn_wmma_f32_16x16x32_f16(
            false, a.v, false, b.v, (short)0, acc, false, false);
    }
    {
        const float bias = b1[ncol];
#pragma unroll
        for (int r = 0; r < 8; ++r) {
            float v = acc[r] + bias;                    // D: M = r + 8*hi, N = ncol
            sT[r + hi * 8][ncol] = (_Float16)fmaxf(v, 0.0f);
        }
    }
    __syncthreads();

    // ---- swap W2 into the weight buffer ----------------------------------
    for (int i = tid; i < DFEAT * DFEAT; i += 256) {
        int k = i >> 7, n = i & 127;
        sW[n][k] = (_Float16)w2[i];
    }
    __syncthreads();

    // ---- stage 2: out = T @ W2 + b2 --------------------------------------
    v8f acc2 = {};
#pragma unroll
    for (int kb = 0; kb < 4; ++kb) {
        Frag a, b;
        const int abase = kb * 32 + hi * 8;
        a.h[0] = *(const v8h*)&sT[mrow][abase];
        a.h[1] = *(const v8h*)&sT[mrow][abase + 16];
        const int bbase = kb * 32 + hi * 8;
        b.h[0] = *(const v8h*)&sW[ncol][bbase];
        b.h[1] = *(const v8h*)&sW[ncol][bbase + 16];
        acc2 = __builtin_amdgcn_wmma_f32_16x16x32_f16(
            false, a.v, false, b.v, (short)0, acc2, false, false);
    }
    {
        const float bias = b2[ncol];
#pragma unroll
        for (int r = 0; r < 8; ++r) {
            int row = m0 + r + hi * 8;
            if (row < nodes)
                __builtin_nontemporal_store(acc2[r] + bias,
                                            out + (size_t)row * DFEAT + ncol);
        }
    }
}

// ---------------------------------------------------------------------------
extern "C" void kernel_launch(void* const* d_in, const int* in_sizes, int n_in,
                              void* d_out, int out_size, void* d_ws, size_t ws_size,
                              hipStream_t stream)
{
    const float* x   = (const float*)d_in[0];   // [N,128] f32
    const int*   ei  = (const int*)  d_in[1];   // [2,E]   int32
    const float* ea  = (const float*)d_in[2];   // [E,128] f32
    const float* w1  = (const float*)d_in[3];
    const float* b1  = (const float*)d_in[4];
    const float* w2  = (const float*)d_in[5];
    const float* b2  = (const float*)d_in[6];
    float* out = (float*)d_out;

    const int nodes  = in_sizes[0] / DFEAT;
    const int nedges = in_sizes[1] / 2;

    float* agg = (float*)d_ws;                  // [N,128] f32 scratch (51.2 MB)

    const int n4 = nodes * (DFEAT / 4);
    gine_init_kernel<<<(n4 + 255) / 256, 256, 0, stream>>>(
        (const float4*)x, (float4*)agg, n4);

    gine_edge_kernel<<<(nedges + 7) / 8, 256, 0, stream>>>(
        x, ei, ea, agg, nedges);

    gine_mlp_kernel<<<(nodes + 15) / 16, 256, 0, stream>>>(
        agg, w1, b1, w2, b2, out, nodes);
}
